// MylossNet_42494406426744
// MI455X (gfx1250) — matrix operand order repaired and checked by reading, not compiled, not run on hardware
//
#include <hip/hip_runtime.h>

typedef _Float16 v16h __attribute__((ext_vector_type(16)));
typedef _Float16 v8h  __attribute__((ext_vector_type(8)));
typedef float    v8f  __attribute__((ext_vector_type(8)));

#define B_HALF 4096
#define DIMK   128

// ---- CDNA5 async global->LDS DMA: two b128 packets sharing base regs.
// INST_OFFSET is added to BOTH the LDS and global address (ISA 08 §4.4),
// and our LDS/global offsets advance in lockstep, so offset: covers both.
__device__ __forceinline__ void async_ld_2x128(uint32_t loff, const void* g) {
    asm volatile("global_load_async_to_lds_b128 %0, %1, off\n\t"
                 "global_load_async_to_lds_b128 %0, %1, off offset:16"
                 :: "v"(loff), "v"(g)
                 : "memory");
}
__device__ __forceinline__ void wait_async0() {
    asm volatile("s_wait_asynccnt 0x0" ::: "memory");
}

// ---------------- zero kernel ----------------
__global__ void zero_kernel(float* p, int n) {
    int i = blockIdx.x * blockDim.x + threadIdx.x;
    if (i < n) p[i] = 0.0f;
}

// ------------- normalize + diag dots -------------
// grid (4096, 3), block 128. Row i of half1 & half2 of one tensor.
__global__ void norm_kernel(const float* __restrict__ X, const float* __restrict__ Y,
                            const float* __restrict__ Z,
                            _Float16* Hx1, _Float16* Hx2,
                            _Float16* Hy1, _Float16* Hy2,
                            _Float16* Hz1, _Float16* Hz2,
                            float* dx, float* dy, float* dz) {
    const int i = blockIdx.x, t = blockIdx.y, tid = threadIdx.x;
    const float* In; _Float16* O1; _Float16* O2; float* D;
    if (t == 0)      { In = X; O1 = Hx1; O2 = Hx2; D = dx; }
    else if (t == 1) { In = Y; O1 = Hy1; O2 = Hy2; D = dy; }
    else             { In = Z; O1 = Hz1; O2 = Hz2; D = dz; }

    float v1 = In[(size_t)i * DIMK + tid];
    float v2 = In[(size_t)(B_HALF + i) * DIMK + tid];

    __shared__ float s1[128], s2[128], s3[128];
    s1[tid] = v1 * v1; s2[tid] = v2 * v2; s3[tid] = v1 * v2;
    __syncthreads();
    for (int s = 64; s > 0; s >>= 1) {
        if (tid < s) { s1[tid] += s1[tid + s]; s2[tid] += s2[tid + s]; s3[tid] += s3[tid + s]; }
        __syncthreads();
    }
    float n1 = fmaxf(sqrtf(s1[0]), 1e-12f);
    float n2 = fmaxf(sqrtf(s2[0]), 1e-12f);
    O1[(size_t)i * DIMK + tid] = (_Float16)(v1 / n1);
    O2[(size_t)i * DIMK + tid] = (_Float16)(v2 / n2);
    if (tid == 0) D[i] = s3[0] / (n1 * n2);
}

// ------------- fused WMMA GEMM + exp + row/col sums -------------
// C = sum_t A_t @ B_t^T  (4096x4096, K=128 per tensor), then
// rowsum[i] += sum_j exp(C[i][j]*10), colsum[j] += sum_i exp(C[i][j]*10)
// grid (32,32) tiles of 128x128, block 256 (8 waves; wave w owns M-rows [16w,16w+16))
// K is processed in 32-wide chunks, double-buffered in LDS; chunk n+1's DMA is
// issued (ASYNCcnt) before computing chunk n, so tile DMA overlaps WMMA work.
// NT/DOCOL are compile-time so the chunk loop fully unrolls (no per-iteration
// pointer selects; scheduler can interleave DS loads/WMMAs across chunks).
template <int NT, bool DOCOL>
__global__ __launch_bounds__(256) void
gemm_exp_sums(const _Float16* __restrict__ A0, const _Float16* __restrict__ A1,
              const _Float16* __restrict__ A2,
              const _Float16* __restrict__ B0, const _Float16* __restrict__ B1,
              const _Float16* __restrict__ B2,
              float* __restrict__ rowsum, float* __restrict__ colsum) {
    __shared__ __align__(16) _Float16 Ash[2][128][40];  // 32 K-chunk + pad
    __shared__ __align__(16) _Float16 Bsh[2][128][40];
    __shared__ float ldsRow[128];
    __shared__ float ldsCol[128];

    const int tid  = threadIdx.x;
    const int lane = tid & 31, wave = tid >> 5;
    const int l15  = lane & 15, lh = lane >> 4;
    const int i0 = blockIdx.y * 128, j0 = blockIdx.x * 128;

    if (tid < 128) { ldsRow[tid] = 0.0f; ldsCol[tid] = 0.0f; }

    const _Float16* As[3] = {A0, A1, A2};
    const _Float16* Bs[3] = {B0, B1, B2};
    const int nCh = NT * 4;                         // number of 32-wide K chunks

    // staging coords: 128 rows x 32 halfs per matrix / 256 threads = 16 halfs each
    const int srow = tid >> 1, scol = (tid & 1) * 16;

    v8f c[8];
#pragma unroll
    for (int nt = 0; nt < 8; ++nt) c[nt] = (v8f){0,0,0,0,0,0,0,0};

    auto stage = [&](int buf, int ch) {
        const int t = ch >> 2, k0 = (ch & 3) * 32;
        const _Float16* Ag = As[t] + (size_t)(i0 + srow) * DIMK + k0 + scol;
        const _Float16* Bg = Bs[t] + (size_t)(j0 + srow) * DIMK + k0 + scol;
        async_ld_2x128((uint32_t)(uintptr_t)&Ash[buf][srow][scol], Ag);
        async_ld_2x128((uint32_t)(uintptr_t)&Bsh[buf][srow][scol], Bg);
    };

    stage(0, 0);
    wait_async0();
    __syncthreads();

#pragma unroll
    for (int ch = 0; ch < nCh; ++ch) {
        const int buf = ch & 1;
        if (ch + 1 < nCh) stage(buf ^ 1, ch + 1);   // DMA next chunk, overlapped

        // A fragment (16x32 f16): lane half lh -> K {8lh..8lh+7, 16+8lh..16+8lh+7}
        v16h af;
        {
            const _Float16* ap = &Ash[buf][wave * 16 + l15][lh * 8];
            v8h lo = *(const v8h*)ap;
            v8h hi = *(const v8h*)(ap + 16);
#pragma unroll
            for (int e = 0; e < 8; ++e) { af[e] = lo[e]; af[e + 8] = hi[e]; }
        }
        // B fragments (32x16 f16): lane half lh -> K {8lh..8lh+7, 16+8lh..16+8lh+7}, col = l15;
        // prefetch next nt's fragment before issuing current WMMA
        v16h bf;
        {
            const _Float16* bp = &Bsh[buf][l15][lh * 8];
            v8h b0v = *(const v8h*)bp;
            v8h b1v = *(const v8h*)(bp + 16);
#pragma unroll
            for (int e = 0; e < 8; ++e) { bf[e] = b0v[e]; bf[e + 8] = b1v[e]; }
        }
#pragma unroll
        for (int nt = 0; nt < 8; ++nt) {
            v16h cur = bf;
            if (nt < 7) {
                const _Float16* bp = &Bsh[buf][(nt + 1) * 16 + l15][lh * 8];
                v8h b0v = *(const v8h*)bp;
                v8h b1v = *(const v8h*)(bp + 16);
#pragma unroll
                for (int e = 0; e < 8; ++e) { bf[e] = b0v[e]; bf[e + 8] = b1v[e]; }
            }
            c[nt] = __builtin_amdgcn_wmma_f32_16x16x32_f16(
                false, af, false, cur, (short)0, c[nt], false, false);
        }

        wait_async0();          // next chunk's DMA (issued above) has landed
        __syncthreads();        // everyone done reading buf; next tile visible
    }

    // epilogue: e = exp(c/T) = exp2(c * 10*log2(e)); fuse row & col reductions
    const float SC = 14.4269504088896340736f;
    float rAcc[8];
#pragma unroll
    for (int r = 0; r < 8; ++r) rAcc[r] = 0.0f;

#pragma unroll
    for (int nt = 0; nt < 8; ++nt) {
        float csum = 0.0f;
#pragma unroll
        for (int r = 0; r < 8; ++r) {
            float ev = exp2f(c[nt][r] * SC);
            rAcc[r] += ev;          // partial row sum (this lane's column)
            csum += ev;             // partial col sum over 8 M-rows
        }
        if (DOCOL) {
            csum += __shfl_xor(csum, 16, 32);      // combine M 0-7 with 8-15
            if (lh == 0) atomicAdd(&ldsCol[nt * 16 + l15], csum);
        }
    }
    // reduce row partials across the 16 lanes of each half
#pragma unroll
    for (int r = 0; r < 8; ++r) {
        float v = rAcc[r];
        v += __shfl_xor(v, 1, 32);
        v += __shfl_xor(v, 2, 32);
        v += __shfl_xor(v, 4, 32);
        v += __shfl_xor(v, 8, 32);
        if (l15 == 0) atomicAdd(&ldsRow[wave * 16 + lh * 8 + r], v);
    }
    __syncthreads();
    if (tid < 128) {
        atomicAdd(&rowsum[i0 + tid], ldsRow[tid]);
        if (DOCOL) atomicAdd(&colsum[j0 + tid], ldsCol[tid]);
    }
}

// ------------- final scalar reduction -------------
__global__ void final_kernel(const float* dx, const float* dy, const float* dz,
                             const float* SxyR, const float* SxyC,
                             const float* SxzR, const float* SxzC,
                             const float* SyzR, const float* SyzC,
                             const float* SvR, float* out) {
    const int tid = threadIdx.x;
    float acc = 0.0f;
    for (int i = tid; i < B_HALF; i += 256) {
        float l1 = -dx[i] * 10.0f + logf(SxyR[i] + SxzR[i]);
        float l2 = -dy[i] * 10.0f + logf(SxyC[i] + SyzR[i]);
        float l3 = -dz[i] * 10.0f + logf(SxzC[i] + SyzC[i]);
        float l4 = -(dx[i] + dy[i] + dz[i]) * 10.0f + logf(SvR[i]);
        acc += l1 + l2 + l3 + l4;
    }
    __shared__ float red[256];
    red[tid] = acc;
    __syncthreads();
    for (int s = 128; s > 0; s >>= 1) {
        if (tid < s) red[tid] += red[tid + s];
        __syncthreads();
    }
    if (tid == 0) out[0] = red[0] / (float)B_HALF;
}

extern "C" void kernel_launch(void* const* d_in, const int* in_sizes, int n_in,
                              void* d_out, int out_size, void* d_ws, size_t ws_size,
                              hipStream_t stream) {
    const float* x = (const float*)d_in[0];
    const float* y = (const float*)d_in[1];
    const float* z = (const float*)d_in[2];
    float* out = (float*)d_out;

    char* ws = (char*)d_ws;
    const size_t HB = (size_t)B_HALF * DIMK * sizeof(_Float16); // 1 MB per half-tensor
    _Float16* Hx1 = (_Float16*)(ws + 0 * HB);
    _Float16* Hx2 = (_Float16*)(ws + 1 * HB);
    _Float16* Hy1 = (_Float16*)(ws + 2 * HB);
    _Float16* Hy2 = (_Float16*)(ws + 3 * HB);
    _Float16* Hz1 = (_Float16*)(ws + 4 * HB);
    _Float16* Hz2 = (_Float16*)(ws + 5 * HB);
    float* fbase = (float*)(ws + 6 * HB);
    float* dx   = fbase + 0 * B_HALF;
    float* dy   = fbase + 1 * B_HALF;
    float* dz   = fbase + 2 * B_HALF;
    float* SxyR = fbase + 3 * B_HALF;   // 7 contiguous sum arrays
    float* SxyC = fbase + 4 * B_HALF;
    float* SxzR = fbase + 5 * B_HALF;
    float* SxzC = fbase + 6 * B_HALF;
    float* SyzR = fbase + 7 * B_HALF;
    float* SyzC = fbase + 8 * B_HALF;
    float* SvR  = fbase + 9 * B_HALF;

    zero_kernel<<<(7 * B_HALF + 255) / 256, 256, 0, stream>>>(SxyR, 7 * B_HALF);
    norm_kernel<<<dim3(B_HALF, 3), 128, 0, stream>>>(x, y, z, Hx1, Hx2, Hy1, Hy2,
                                                     Hz1, Hz2, dx, dy, dz);

    dim3 g(32, 32), b(256);
    // cross GEMMs among first halves: rows+cols of exp()
    gemm_exp_sums<1, true><<<g, b, 0, stream>>>(Hx1, nullptr, nullptr,
                                                Hy1, nullptr, nullptr, SxyR, SxyC);
    gemm_exp_sums<1, true><<<g, b, 0, stream>>>(Hx1, nullptr, nullptr,
                                                Hz1, nullptr, nullptr, SxzR, SxzC);
    gemm_exp_sums<1, true><<<g, b, 0, stream>>>(Hy1, nullptr, nullptr,
                                                Hz1, nullptr, nullptr, SyzR, SyzC);
    // view GEMM: xyz1 @ xyz2^T = sum of three K=128 GEMMs; rows only
    gemm_exp_sums<3, false><<<g, b, 0, stream>>>(Hx1, Hy1, Hz1,
                                                 Hx2, Hy2, Hz2, SvR, nullptr);

    final_kernel<<<1, 256, 0, stream>>>(dx, dy, dz, SxyR, SxyC, SxzR, SxzC,
                                        SyzR, SyzC, SvR, out);
}
